// IncompleteFeatSimulator_17179869326
// MI455X (gfx1250) — matrix operand order repaired and checked by reading, not compiled, not run on hardware
//
#include <hip/hip_runtime.h>
#include <hip/hip_bf16.h>
#include <stdint.h>

// ---------------------------------------------------------------------------
// MI455X (gfx1250): 4 chained bf16 WMMA GEMMs with fused routing selects.
// wave32, v_wmma_f32_16x16x32_bf16, double-buffered LDS fed by
// GLOBAL_LOAD_ASYNC_TO_LDS_B128, plus register-level double buffering of the
// B fragments so ds_load latency overlaps WMMA execution.
// ---------------------------------------------------------------------------

typedef __bf16 bf16_t;
typedef __bf16 v16bf __attribute__((ext_vector_type(16)));
typedef float  v8f   __attribute__((ext_vector_type(8)));

#define DIMK   1024
#define NROWS  65536
#define BM     128
#define BN     128
#define BK     64          // K per LDS stage (two WMMA K-steps of 32)

enum { EP_BF16 = 0, EP_SEL_BF16 = 1, EP_F32_FINAL = 2 };

__device__ __forceinline__ unsigned short f2bf(float f) {
  unsigned u = __builtin_bit_cast(unsigned, f);
  unsigned r = u + 0x7FFFu + ((u >> 16) & 1u);   // round-to-nearest-even
  return (unsigned short)(r >> 16);
}

union FragU { v16bf v; uint4 q[2]; };

// async copy: 16 bytes global -> LDS, tracked with ASYNCcnt (CDNA5)
__device__ __forceinline__ void async_ld16(unsigned ldsoff, const void* g) {
  unsigned long long ga = (unsigned long long)(size_t)g;
  asm volatile("global_load_async_to_lds_b128 %0, %1, off"
               :: "v"(ldsoff), "v"(ga) : "memory");
}
__device__ __forceinline__ void wait_async0() {
  asm volatile("s_wait_asynccnt 0x0" ::: "memory");
}

// ---------------------------------------------------------------------------
// Routing: levels[i] = max(argmax3(x_angle[i]), argmax3(y_angle[i]))
// ---------------------------------------------------------------------------
__global__ __launch_bounds__(256) void route_kernel(
    const float* __restrict__ xa, const float* __restrict__ ya,
    unsigned char* __restrict__ levels, int n) {
  int i = blockIdx.x * blockDim.x + threadIdx.x;
  if (i >= n) return;
  const float* a = xa + (size_t)i * 3;
  int la = 0; float ma = a[0];
  if (a[1] > ma) { ma = a[1]; la = 1; }
  if (a[2] > ma) { ma = a[2]; la = 2; }
  const float* b = ya + (size_t)i * 3;
  int lb = 0; float mb = b[0];
  if (b[1] > mb) { mb = b[1]; lb = 1; }
  if (b[2] > mb) { mb = b[2]; lb = 2; }
  levels[i] = (unsigned char)(la > lb ? la : lb);
}

// ---------------------------------------------------------------------------
// fp32 -> bf16 weight conversion (4 floats / thread)
// ---------------------------------------------------------------------------
__global__ __launch_bounds__(256) void convert_f32_bf16(
    const float* __restrict__ src, bf16_t* __restrict__ dst) {
  size_t i = ((size_t)blockIdx.x * blockDim.x + threadIdx.x) * 4;
  float4 f = *reinterpret_cast<const float4*>(src + i);
  uint2 p;
  p.x = (unsigned)f2bf(f.x) | ((unsigned)f2bf(f.y) << 16);
  p.y = (unsigned)f2bf(f.z) | ((unsigned)f2bf(f.w) << 16);
  *reinterpret_cast<uint2*>(dst + i) = p;
}

// ---------------------------------------------------------------------------
// Out[row, col] = A[row, :] . W[col, :] + bias[col]   (A @ W^T + b)
// Block: 256 threads (8 waves). Block tile 128x128; each wave: 16 rows x 128 cols.
// ---------------------------------------------------------------------------
template<bool AF32, int EP>
__global__ __launch_bounds__(256) void gemm_bias_kernel(
    const bf16_t* __restrict__ Abf, const float* __restrict__ Af32,
    const bf16_t* __restrict__ Wb, const float* __restrict__ bias,
    const unsigned char* __restrict__ levels, const float* __restrict__ Xf32,
    bf16_t* __restrict__ outB, float* __restrict__ outF) {
  __shared__ bf16_t As[2][BM * BK];   // 16 KB per buffer
  __shared__ bf16_t Ws[2][BN * BK];   // 16 KB per buffer

  const int tid     = threadIdx.x;
  const int lane    = tid & 31;
  const int wave    = tid >> 5;
  const int colBase = blockIdx.x * BN;
  const int rowBase = blockIdx.y * BM;
  const int nl      = lane & 15;      // lane's N (and A's M) index in a tile
  const int hi      = lane >> 4;      // half-wave selector

  const unsigned asBase = (unsigned)(size_t)&As[0][0];   // LDS byte offsets
  const unsigned wsBase = (unsigned)(size_t)&Ws[0][0];

  v8f acc[8] = {};

  // ---- stage one K-tile (BK columns) into LDS buffer `buf` ----
  auto copy_tile = [&](int k0, int buf) {
    if (AF32) {
      // 128 rows x 64 floats: 2048 float4, 8 per thread; convert to bf16
#pragma unroll
      for (int i = 0; i < 8; ++i) {
        int q = tid + i * 256;
        int row = q >> 4, part = q & 15;
        const float4 f = *reinterpret_cast<const float4*>(
            Af32 + (size_t)(rowBase + row) * DIMK + k0 + part * 4);
        uint2 p;
        p.x = (unsigned)f2bf(f.x) | ((unsigned)f2bf(f.y) << 16);
        p.y = (unsigned)f2bf(f.z) | ((unsigned)f2bf(f.w) << 16);
        *reinterpret_cast<uint2*>(&As[buf][row * BK + part * 4]) = p;
      }
    } else {
      // 128 rows x 64 bf16: 1024 x 16B, 4 per thread, async to LDS
#pragma unroll
      for (int i = 0; i < 4; ++i) {
        int q = tid + i * 256;
        int row = q >> 3, part = q & 7;
        async_ld16(asBase + (unsigned)buf * (BM * BK * 2) + row * (BK * 2) + part * 16,
                   Abf + (size_t)(rowBase + row) * DIMK + k0 + part * 8);
      }
    }
    // W tile: 128 cols x 64 bf16, async to LDS
#pragma unroll
    for (int i = 0; i < 4; ++i) {
      int q = tid + i * 256;
      int j = q >> 3, part = q & 7;
      async_ld16(wsBase + (unsigned)buf * (BN * BK * 2) + j * (BK * 2) + part * 16,
                 Wb + (size_t)(colBase + j) * DIMK + k0 + part * 8);
    }
  };

  copy_tile(0, 0);
  wait_async0();
  __syncthreads();

  int buf = 0;
  for (int k0 = 0; k0 < DIMK; k0 += BK) {
    if (k0 + BK < DIMK) copy_tile(k0 + BK, buf ^ 1);

    const bf16_t* Asb = &As[buf][0];
    const bf16_t* Wsb = &Ws[buf][0];

    // A fragments for both 32-wide K sub-steps (contiguous K order: k = 16*hi + s, matching B)
    FragU a0, a1;
    {
      const int ar = (wave * 16 + nl) * BK;
      a0.q[0] = *reinterpret_cast<const uint4*>(&Asb[ar + hi * 16]);
      a0.q[1] = *reinterpret_cast<const uint4*>(&Asb[ar + hi * 16 + 8]);
      a1.q[0] = *reinterpret_cast<const uint4*>(&Asb[ar + 32 + hi * 16]);
      a1.q[1] = *reinterpret_cast<const uint4*>(&Asb[ar + 32 + hi * 16 + 8]);
    }

    // B fragments: register double buffer -> loads for tile c+1 stay in
    // flight while the WMMAs for tile c execute (s_wait_dscnt <= 4).
    FragU b0[2], b1[2];
    {
      const int jl = nl * BK;
      b0[0].q[0] = *reinterpret_cast<const uint4*>(&Wsb[jl + hi * 16]);
      b0[0].q[1] = *reinterpret_cast<const uint4*>(&Wsb[jl + hi * 16 + 8]);
      b1[0].q[0] = *reinterpret_cast<const uint4*>(&Wsb[jl + 32 + hi * 16]);
      b1[0].q[1] = *reinterpret_cast<const uint4*>(&Wsb[jl + 32 + hi * 16 + 8]);
    }
#pragma unroll
    for (int c = 0; c < 8; ++c) {
      const int cur = c & 1, nxt = cur ^ 1;
      if (c < 7) {
        const int jl = ((c + 1) * 16 + nl) * BK;
        b0[nxt].q[0] = *reinterpret_cast<const uint4*>(&Wsb[jl + hi * 16]);
        b0[nxt].q[1] = *reinterpret_cast<const uint4*>(&Wsb[jl + hi * 16 + 8]);
        b1[nxt].q[0] = *reinterpret_cast<const uint4*>(&Wsb[jl + 32 + hi * 16]);
        b1[nxt].q[1] = *reinterpret_cast<const uint4*>(&Wsb[jl + 32 + hi * 16 + 8]);
      }
      acc[c] = __builtin_amdgcn_wmma_f32_16x16x32_bf16(
          false, a0.v, false, b0[cur].v, (short)0, acc[c], false, false);
      acc[c] = __builtin_amdgcn_wmma_f32_16x16x32_bf16(
          false, a1.v, false, b1[cur].v, (short)0, acc[c], false, false);
    }

    wait_async0();      // our async writes to buf^1 have landed
    __syncthreads();    // everyone done reading buf, writes visible
    buf ^= 1;
  }

  // -------- epilogue: bias + routing select --------
  float bcol[8];
#pragma unroll
  for (int c = 0; c < 8; ++c) bcol[c] = bias[colBase + c * 16 + nl];

#pragma unroll
  for (int r = 0; r < 8; ++r) {
    const int row = rowBase + wave * 16 + hi * 8 + r;   // C/D: VGPR r -> M=r+8*hi
    int lvl = 0;
    if (EP != EP_BF16) lvl = (int)levels[row];
#pragma unroll
    for (int c = 0; c < 8; ++c) {
      const int col = colBase + c * 16 + nl;
      const float v = acc[c][r] + bcol[c];
      const size_t idx = (size_t)row * DIMK + col;
      if (EP == EP_BF16) {
        *reinterpret_cast<unsigned short*>(&outB[idx]) = f2bf(v);
      } else if (EP == EP_SEL_BF16) {
        const float pick = (lvl == 2) ? v : Xf32[idx];
        *reinterpret_cast<unsigned short*>(&outB[idx]) = f2bf(pick);
      } else {
        outF[idx] = (lvl >= 1) ? v : Xf32[idx];
      }
    }
  }
}

// ---------------------------------------------------------------------------
extern "C" void kernel_launch(void* const* d_in, const int* in_sizes, int n_in,
                              void* d_out, int out_size, void* d_ws, size_t ws_size,
                              hipStream_t stream) {
  (void)in_sizes; (void)n_in; (void)out_size; (void)ws_size;
  const float* x_feat  = (const float*)d_in[0];
  const float* x_angle = (const float*)d_in[1];
  const float* y_angle = (const float*)d_in[2];
  const float* W[4] = {(const float*)d_in[3], (const float*)d_in[5],
                       (const float*)d_in[7], (const float*)d_in[9]};
  const float* b[4] = {(const float*)d_in[4], (const float*)d_in[6],
                       (const float*)d_in[8], (const float*)d_in[10]};
  float* out = (float*)d_out;

  // workspace carve-up (~272 MB)
  char* ws = (char*)d_ws;
  size_t off = 0;
  unsigned char* levels = (unsigned char*)(ws + off);
  off += NROWS;               off = (off + 255) & ~(size_t)255;
  bf16_t* Wbf = (bf16_t*)(ws + off);
  off += (size_t)4 * DIMK * DIMK * sizeof(bf16_t);  off = (off + 255) & ~(size_t)255;
  bf16_t* buf1 = (bf16_t*)(ws + off);
  off += (size_t)NROWS * DIMK * sizeof(bf16_t);     off = (off + 255) & ~(size_t)255;
  bf16_t* buf2 = (bf16_t*)(ws + off);

  // 1) weights -> bf16
  for (int i = 0; i < 4; ++i) {
    convert_f32_bf16<<<(DIMK * DIMK) / (256 * 4), 256, 0, stream>>>(
        W[i], Wbf + (size_t)i * DIMK * DIMK);
  }
  // 2) routing levels
  route_kernel<<<(NROWS + 255) / 256, 256, 0, stream>>>(x_angle, y_angle, levels, NROWS);

  dim3 grid(DIMK / BN, NROWS / BM);   // (8, 512)
  // 3) T1 = X @ W1^T + b1          (fp32 input, converted inline)
  gemm_bias_kernel<true, EP_BF16><<<grid, 256, 0, stream>>>(
      nullptr, x_feat, Wbf + (size_t)0 * DIMK * DIMK, b[0],
      nullptr, nullptr, buf1, nullptr);
  // 4) Y  = l3 ? T1 @ W2^T + b2 : X
  gemm_bias_kernel<false, EP_SEL_BF16><<<grid, 256, 0, stream>>>(
      buf1, nullptr, Wbf + (size_t)1 * DIMK * DIMK, b[1],
      levels, x_feat, buf2, nullptr);
  // 5) T3 = Y @ W3^T + b3
  gemm_bias_kernel<false, EP_BF16><<<grid, 256, 0, stream>>>(
      buf2, nullptr, Wbf + (size_t)2 * DIMK * DIMK, b[2],
      nullptr, nullptr, buf1, nullptr);
  // 6) out = level>=1 ? T3 @ W4^T + b4 : X   (fp32 output)
  gemm_bias_kernel<false, EP_F32_FINAL><<<grid, 256, 0, stream>>>(
      buf1, nullptr, Wbf + (size_t)3 * DIMK * DIMK, b[3],
      levels, x_feat, nullptr, out);
}
